// TinyAttention_26465588478026
// MI455X (gfx1250) — matrix operand order repaired and checked by reading, not compiled, not run on hardware
//
#include <hip/hip_runtime.h>
#include <stdint.h>

#define HIDDEN   2048
#define HEADS    16
#define HEAD_DIM 128
#define BATCH    4
#define SEQ      2048
#define QKV_LD   (3 * HIDDEN)        // 6144
#define ROWS     (BATCH * SEQ)       // 8192

typedef __attribute__((ext_vector_type(16))) __bf16 v16bf;
typedef __attribute__((ext_vector_type(8)))  float  v8f;
typedef __attribute__((ext_vector_type(4)))  unsigned int v4u;
typedef __attribute__((ext_vector_type(8)))  int    v8i;
typedef __attribute__((ext_vector_type(4)))  int    v4i;

union FragU { uint4 u[2]; v16bf v; };
union U4x8  { uint4 u; uint16_t s[8]; };

__device__ __forceinline__ uint16_t f32_to_bf16(float f) {
  union { float f; uint32_t u; } x; x.f = f;
  uint32_t r = x.u + 0x7FFFu + ((x.u >> 16) & 1u);   // round-to-nearest-even
  return (uint16_t)(r >> 16);
}
__device__ __forceinline__ float bf16_to_f32(uint16_t h) {
  union { uint32_t u; float f; } x; x.u = ((uint32_t)h) << 16;
  return x.f;
}
__device__ __forceinline__ v8f vzero8() {
  v8f v;
#pragma unroll
  for (int i = 0; i < 8; ++i) v[i] = 0.0f;
  return v;
}

// A-matrix 16x32 bf16 fragment (row-major source, ld in elements).
// ISA layout: lanes 0-15 hold row m, K {0..7} and {16..23}; lanes 16-31 hold K {8..15},{24..31}.
__device__ __forceinline__ v16bf load_a_frag(const uint16_t* base, int ld, int k0) {
  const int lane = threadIdx.x & 31;
  const int mrow = lane & 15;
  const int hi   = lane >> 4;
  const uint16_t* p = base + (size_t)mrow * ld + k0 + hi * 8;
  FragU f;
  f.u[0] = *(const uint4*)(p);
  f.u[1] = *(const uint4*)(p + 16);
  return f.v;
}

// B-matrix 32x16 bf16 fragment. Source stored "N x K" row-major (i.e. B = srcT),
// so lane n reads K values from row n (lanes 0-15: K {0..7},{16..23}; 16-31: K {8..15},{24..31}).
__device__ __forceinline__ v16bf load_b_frag(const uint16_t* base, int ld, int k0) {
  const int lane = threadIdx.x & 31;
  const int ncol = lane & 15;
  const int hi   = lane >> 4;
  const uint16_t* p = base + (size_t)ncol * ld + k0 + hi * 8;
  FragU f;
  f.u[0] = *(const uint4*)(p);
  f.u[1] = *(const uint4*)(p + 16);
  return f.v;
}

// Tensor Data Mover: DMA one 32x128 bf16 K tile (rows strided QKV_LD) into LDS,
// padding 4 dwords after every 64 dwords -> 136-element LDS rows (bank spread).
// D# per CDNA5 ISA; 2-D tensor, groups 2/3 zero.
// This toolchain exposes the 6-arg builtin: (v4u g0, v8i g1, v4i g2, v4i g3, v8i g4, i32 cpol).
__device__ __forceinline__ void tdm_load_k_tile(const uint16_t* gsrc, uint32_t lds_off) {
  uint64_t ga = (uint64_t)(uintptr_t)gsrc;
  v4u g0;
  g0[0] = 1u;                                   // count=1 user descriptor
  g0[1] = lds_off;                              // lds_addr (bytes)
  g0[2] = (uint32_t)ga;                         // global_addr[31:0]
  g0[3] = (uint32_t)(ga >> 32) | 0x80000000u;   // global_addr[56:32] | type=2 (bits 127:126)
  v8i g1;
  g1[0] = (int)((1u << 16) | (1u << 20) | (5u << 22) | (3u << 25));
  //            data_size=2B  pad_enable  pad_int=64dw  pad_amt=4dw
  g1[1] = (int)((uint32_t)QKV_LD << 16);        // tensor_dim0 = 6144 (bits 79:48 lo half)
  g1[2] = (int)((uint32_t)ROWS << 16);          // tensor_dim0 hi=0 | tensor_dim1 = 8192 lo
  g1[3] = (int)((uint32_t)HEAD_DIM << 16);      // tensor_dim1 hi=0 | tile_dim0 = 128
  g1[4] = 32;                                   // tile_dim1 = 32 | tile_dim2 = 0
  g1[5] = QKV_LD;                               // tensor_dim0_stride = 6144 (lo 32)
  g1[6] = 0;                                    // stride0 hi | dim1_stride lo
  g1[7] = 0;
  v4i z4 = {0, 0, 0, 0};
  v8i z8 = {0, 0, 0, 0, 0, 0, 0, 0};
  __builtin_amdgcn_tensor_load_to_lds(g0, g1, z4, z4, z8, 0);
}

// ---------------------------------------------------------------- conversion
__global__ __launch_bounds__(256) void cvt_f32_bf16(const float4* __restrict__ src,
                                                    ushort4* __restrict__ dst, int n4) {
  int i = blockIdx.x * 256 + threadIdx.x;
  if (i < n4) {
    float4 f = src[i];
    ushort4 o;
    o.x = f32_to_bf16(f.x); o.y = f32_to_bf16(f.y);
    o.z = f32_to_bf16(f.z); o.w = f32_to_bf16(f.w);
    dst[i] = o;
  }
}

// ---------------------------------------------------------------- GEMM: C = A * B^T
// A: M x K bf16 row-major.  Bm: N x K bf16 row-major.  C: M x N (f32 or bf16).
// Wave tile 64x32 (4x2 accumulators), block = 8 waves = 256x64 tile.
// Ping-pong fragment sets: next k-step's global loads in flight during WMMAs.
template <bool OUTF32>
__global__ __launch_bounds__(256) void gemm_bt(const uint16_t* __restrict__ A,
                                               const uint16_t* __restrict__ Bm,
                                               void* __restrict__ Cv,
                                               int K, int lda, int ldb, int ldc) {
  const int lane = threadIdx.x & 31;
  const int wid  = threadIdx.x >> 5;
  const int wm   = wid >> 1;            // 0..3 along M
  const int wn   = wid & 1;             // 0..1 along N
  const int nlo  = lane & 15;
  const int hi   = lane >> 4;
  const size_t row0 = (size_t)blockIdx.y * 256 + wm * 64;
  const int    col0 = blockIdx.x * 64 + wn * 32;

  v8f acc[4][2];
#pragma unroll
  for (int i = 0; i < 4; ++i)
#pragma unroll
    for (int j = 0; j < 2; ++j) acc[i][j] = vzero8();

  v16bf a0[4], b0[2], a1[4], b1[2];
#pragma unroll
  for (int i = 0; i < 4; ++i) a0[i] = load_a_frag(A + (row0 + 16 * i) * (size_t)lda, lda, 0);
#pragma unroll
  for (int j = 0; j < 2; ++j) b0[j] = load_b_frag(Bm + (size_t)(col0 + 16 * j) * ldb, ldb, 0);

  for (int k0 = 0; k0 < K; k0 += 64) {
#pragma unroll
    for (int i = 0; i < 4; ++i) a1[i] = load_a_frag(A + (row0 + 16 * i) * (size_t)lda, lda, k0 + 32);
#pragma unroll
    for (int j = 0; j < 2; ++j) b1[j] = load_b_frag(Bm + (size_t)(col0 + 16 * j) * ldb, ldb, k0 + 32);
#pragma unroll
    for (int i = 0; i < 4; ++i)
#pragma unroll
      for (int j = 0; j < 2; ++j)
        acc[i][j] = __builtin_amdgcn_wmma_f32_16x16x32_bf16(
            false, a0[i], false, b0[j], (short)0, acc[i][j], false, false);
    if (k0 + 64 < K) {
#pragma unroll
      for (int i = 0; i < 4; ++i) a0[i] = load_a_frag(A + (row0 + 16 * i) * (size_t)lda, lda, k0 + 64);
#pragma unroll
      for (int j = 0; j < 2; ++j) b0[j] = load_b_frag(Bm + (size_t)(col0 + 16 * j) * ldb, ldb, k0 + 64);
    }
#pragma unroll
    for (int i = 0; i < 4; ++i)
#pragma unroll
      for (int j = 0; j < 2; ++j)
        acc[i][j] = __builtin_amdgcn_wmma_f32_16x16x32_bf16(
            false, a1[i], false, b1[j], (short)0, acc[i][j], false, false);
  }

#pragma unroll
  for (int i = 0; i < 4; ++i)
#pragma unroll
    for (int j = 0; j < 2; ++j)
#pragma unroll
      for (int rr = 0; rr < 8; ++rr) {
        size_t row = row0 + 16 * i + rr + hi * 8;     // C layout: VGPR rr -> M rr (+8 for hi half)
        int    col = col0 + 16 * j + nlo;
        if (OUTF32) ((float*)Cv)[row * ldc + col] = acc[i][j][rr];
        else        ((uint16_t*)Cv)[row * ldc + col] = f32_to_bf16(acc[i][j][rr]);
      }
}

// ---------------------------------------------------------------- RoPE (in place on bf16 qkv)
__global__ __launch_bounds__(256) void rope_kernel(uint16_t* __restrict__ qkv,
                                                   const float* __restrict__ cosb,
                                                   const float* __restrict__ sinb) {
  int idx = blockIdx.x * 256 + threadIdx.x;
  const int d   = idx & 63;  idx >>= 6;
  const int h   = idx & 15;  idx >>= 4;
  const int isk = idx & 1;   idx >>= 1;
  const int row = idx;                       // b*SEQ + l
  const int l   = row & (SEQ - 1);

  uint16_t* p = qkv + (size_t)row * QKV_LD + (isk ? HIDDEN : 0) + h * HEAD_DIM + d;
  float x1 = bf16_to_f32(p[0]);
  float x2 = bf16_to_f32(p[64]);
  float c1 = cosb[l * HEAD_DIM + d],      s1 = sinb[l * HEAD_DIM + d];
  float c2 = cosb[l * HEAD_DIM + d + 64], s2 = sinb[l * HEAD_DIM + d + 64];
  float o1 = x1 * c1 - x2 * s1;
  float o2 = x2 * c2 + x1 * s2;
  if (!isk) { o1 *= 0.08838834764831845f; o2 *= 0.08838834764831845f; }  // 1/sqrt(128)
  p[0]  = f32_to_bf16(o1);
  p[64] = f32_to_bf16(o2);
}

// ---------------------------------------------------------------- flash attention
// grid = (SEQ/64, BATCH*HEADS); 128 threads = 4 waves; each wave owns 16 query rows.
// K tiles: TDM tensor_load_to_lds, double-buffered, issued one tile ahead during the
// compute phase (TENSORcnt-tracked; waited at the next staging phase).
// V tiles: pipelined through registers, transposed at LDS store.
__global__ __launch_bounds__(128) void attn_fa(const uint16_t* __restrict__ qkv,
                                               uint16_t* __restrict__ outb) {
  __shared__ uint16_t Kt[2][32][136];  // K tile ping-pong; TDM pads rows to 136 elems
  __shared__ uint16_t Vt[128][40];     // V tile transposed (dim, key)
  __shared__ uint16_t Pt[4][16][40];   // per-wave P re-layout buffer (C -> A layout)

  const int tid  = threadIdx.x;
  const int lane = tid & 31;
  const int wid  = tid >> 5;
  const int nlo  = lane & 15;
  const int hi   = lane >> 4;

  const int b = blockIdx.y >> 4;
  const int h = blockIdx.y & 15;
  const size_t rowbase = (size_t)b * SEQ;
  const uint16_t* qptr = qkv + (rowbase + blockIdx.x * 64 + wid * 16) * QKV_LD + h * HEAD_DIM;

  v16bf aq[4];
#pragma unroll
  for (int kc = 0; kc < 4; ++kc) aq[kc] = load_a_frag(qptr, QKV_LD, kc * 32);

  v8f o[8];
#pragma unroll
  for (int n = 0; n < 8; ++n) o[n] = vzero8();
  float m[8], l[8];
#pragma unroll
  for (int rr = 0; rr < 8; ++rr) { m[rr] = -1e30f; l[rr] = 0.0f; }

  const uint16_t* Kbase = qkv + rowbase * QKV_LD + HIDDEN     + h * HEAD_DIM;
  const uint16_t* Vbase = qkv + rowbase * QKV_LD + 2 * HIDDEN + h * HEAD_DIM;

  const int sr = tid >> 2;           // 0..31: staging row (key index in tile)
  const int sc = (tid & 3) * 32;     // 0,32,64,96: staging column base

  // prologue: TDM kicks off K tile 0; V tile 0 into registers
  if (wid == 0)
    tdm_load_k_tile(Kbase, (uint32_t)(uintptr_t)&Kt[0][0][0]);
  uint4 vreg[4];
#pragma unroll
  for (int ii = 0; ii < 4; ++ii)
    vreg[ii] = *(const uint4*)(Vbase + (size_t)sr * QKV_LD + sc + 8 * ii);

  for (int t = 0; t < SEQ / 32; ++t) {
    const int cur = t & 1;
    // ---- staging phase (previous readers released by loop-end barrier) ----
#pragma unroll
    for (int ii = 0; ii < 4; ++ii) {
      U4x8 u; u.u = vreg[ii];
#pragma unroll
      for (int e = 0; e < 8; ++e) Vt[sc + 8 * ii + e][sr] = u.s[e];   // transpose at store
    }
    if (wid == 0) __builtin_amdgcn_s_wait_tensorcnt(0);   // K tile t landed in Kt[cur]
    __syncthreads();

    // kick off next K tile's DMA + next V tile's register loads: overlap with compute
    if (t + 1 < SEQ / 32) {
      if (wid == 0)
        tdm_load_k_tile(Kbase + (size_t)((t + 1) * 32) * QKV_LD,
                        (uint32_t)(uintptr_t)&Kt[cur ^ 1][0][0]);
      const uint16_t* Vs = Vbase + (size_t)((t + 1) * 32) * QKV_LD;
#pragma unroll
      for (int ii = 0; ii < 4; ++ii)
        vreg[ii] = *(const uint4*)(Vs + (size_t)sr * QKV_LD + sc + 8 * ii);
      if (t + 2 < SEQ / 32)
        __builtin_prefetch(Vbase + (size_t)((t + 2) * 32 + sr) * QKV_LD + sc, 0, 0);
    }

    // ---- S = Q K^T: preload all 8 B frags, then 8 back-to-back WMMAs ----
    v16bf bk[8];
#pragma unroll
    for (int kc = 0; kc < 4; ++kc) {
      bk[2 * kc + 0] = load_b_frag(&Kt[cur][0][0],  136, kc * 32);
      bk[2 * kc + 1] = load_b_frag(&Kt[cur][16][0], 136, kc * 32);
    }
    v8f s0 = vzero8(), s1 = vzero8();
#pragma unroll
    for (int kc = 0; kc < 4; ++kc) {
      s0 = __builtin_amdgcn_wmma_f32_16x16x32_bf16(false, aq[kc], false, bk[2 * kc + 0], (short)0, s0, false, false);
      s1 = __builtin_amdgcn_wmma_f32_16x16x32_bf16(false, aq[kc], false, bk[2 * kc + 1], (short)0, s1, false, false);
    }

    // ---- online softmax (xor-shuffle reductions over 16-lane C-layout halves) ----
#pragma unroll
    for (int rr = 0; rr < 8; ++rr) {
      float mv = fmaxf(s0[rr], s1[rr]);
#pragma unroll
      for (int off = 8; off >= 1; off >>= 1) mv = fmaxf(mv, __shfl_xor(mv, off, 32));
      float mn = fmaxf(m[rr], mv);
      float p0 = __expf(s0[rr] - mn);
      float p1 = __expf(s1[rr] - mn);
      s0[rr] = p0; s1[rr] = p1;
      float rs = p0 + p1;
#pragma unroll
      for (int off = 8; off >= 1; off >>= 1) rs += __shfl_xor(rs, off, 32);
      float alpha = __expf(m[rr] - mn);
      l[rr] = l[rr] * alpha + rs;
      m[rr] = mn;
#pragma unroll
      for (int n = 0; n < 8; ++n) o[n][rr] *= alpha;
    }

    // ---- P: C layout -> bf16 A layout via per-wave LDS round trip ----
#pragma unroll
    for (int rr = 0; rr < 8; ++rr) {
      Pt[wid][rr + hi * 8][nlo]      = f32_to_bf16(s0[rr]);
      Pt[wid][rr + hi * 8][16 + nlo] = f32_to_bf16(s1[rr]);
    }
    v16bf ap = load_a_frag(&Pt[wid][0][0], 40, 0);

    // ---- O += P V: preload all 8 V B-frags, then 8 WMMAs ----
    v16bf bv[8];
#pragma unroll
    for (int n = 0; n < 8; ++n) bv[n] = load_b_frag(&Vt[16 * n][0], 40, 0);
#pragma unroll
    for (int n = 0; n < 8; ++n)
      o[n] = __builtin_amdgcn_wmma_f32_16x16x32_bf16(false, ap, false, bv[n], (short)0, o[n], false, false);
    __syncthreads();
  }

  const size_t orow0 = rowbase + (size_t)blockIdx.x * 64 + wid * 16;
#pragma unroll
  for (int rr = 0; rr < 8; ++rr) {
    float inv = 1.0f / l[rr];
#pragma unroll
    for (int n = 0; n < 8; ++n) {
      size_t row = orow0 + rr + hi * 8;
      int    col = h * HEAD_DIM + 16 * n + nlo;
      outb[row * HIDDEN + col] = f32_to_bf16(o[n][rr] * inv);
    }
  }
}

// ---------------------------------------------------------------- host
extern "C" void kernel_launch(void* const* d_in, const int* in_sizes, int n_in,
                              void* d_out, int out_size, void* d_ws, size_t ws_size,
                              hipStream_t stream) {
  const float* x     = (const float*)d_in[0];
  const float* cosb  = (const float*)d_in[1];
  const float* sinb  = (const float*)d_in[2];
  const float* w_qkv = (const float*)d_in[3];
  const float* w_out = (const float*)d_in[4];
  float* out = (float*)d_out;

  // workspace layout (bf16 = uint16_t); attn output aliases xb (dead after QKV GEMM)
  uint16_t* xb    = (uint16_t*)d_ws;                       // ROWS x HIDDEN      (32 MB)
  uint16_t* wqkvb = xb    + (size_t)ROWS * HIDDEN;         // 3H x H             (24 MB)
  uint16_t* woutb = wqkvb + (size_t)3 * HIDDEN * HIDDEN;   // H x H              ( 8 MB)
  uint16_t* qkvb  = woutb + (size_t)HIDDEN * HIDDEN;       // ROWS x 3H          (96 MB)
  uint16_t* attnb = xb;

  const int nx  = ROWS * HIDDEN;            // 16,777,216
  const int nwq = 3 * HIDDEN * HIDDEN;      // 12,582,912
  const int nwo = HIDDEN * HIDDEN;          //  4,194,304
  cvt_f32_bf16<<<dim3(nx  / 4 / 256), dim3(256), 0, stream>>>((const float4*)x,     (ushort4*)xb,    nx  / 4);
  cvt_f32_bf16<<<dim3(nwq / 4 / 256), dim3(256), 0, stream>>>((const float4*)w_qkv, (ushort4*)wqkvb, nwq / 4);
  cvt_f32_bf16<<<dim3(nwo / 4 / 256), dim3(256), 0, stream>>>((const float4*)w_out, (ushort4*)woutb, nwo / 4);

  // qkv = x @ w_qkv^T   (M=8192, N=6144, K=2048), bf16 out
  gemm_bt<false><<<dim3(QKV_LD / 64, ROWS / 256), dim3(256), 0, stream>>>(
      xb, wqkvb, (void*)qkvb, HIDDEN, HIDDEN, HIDDEN, QKV_LD);

  // RoPE in place on q,k (scale folded into q)
  rope_kernel<<<dim3((ROWS * 2 * HEADS * 64) / 256), dim3(256), 0, stream>>>(qkvb, cosb, sinb);

  // flash attention: 64-query tiles x (b,h)
  attn_fa<<<dim3(SEQ / 64, BATCH * HEADS), dim3(128), 0, stream>>>(qkvb, attnb);

  // out = attn @ w_out^T  (M=8192, N=2048, K=2048), f32 out
  gemm_bt<true><<<dim3(HIDDEN / 64, ROWS / 256), dim3(256), 0, stream>>>(
      attnb, woutb, (void*)out, HIDDEN, HIDDEN, HIDDEN, HIDDEN);
}
